// TransformerBlock_32203664785950
// MI455X (gfx1250) — matrix operand order repaired and checked by reading, not compiled, not run on hardware
//
#include <hip/hip_runtime.h>
#include <math.h>

// ---------------------------------------------------------------------------
// CDNA5 (gfx1250) types & WMMA helpers
// ---------------------------------------------------------------------------
typedef __attribute__((ext_vector_type(16))) __bf16 bf16x16;
typedef __attribute__((ext_vector_type(8)))  __bf16 bf16x8;
typedef __attribute__((ext_vector_type(8)))  float  f32x8;

static __device__ __forceinline__ f32x8 wmma_bf16(bf16x16 a, bf16x16 b, f32x8 c) {
  // D(f32 16x16) = A(bf16 16x32) * B(bf16 32x16) + C
  return __builtin_amdgcn_wmma_f32_16x16x32_bf16(false, a, false, b, (short)0, c,
                                                 false, false);
}

static __device__ __forceinline__ bf16x16 cat8(bf16x8 lo, bf16x8 hi) {
  bf16x16 r;
#pragma unroll
  for (int i = 0; i < 8; ++i) { r[i] = lo[i]; r[i + 8] = hi[i]; }
  return r;
}

static __device__ __forceinline__ float gelu_exact(float v) {
  return 0.5f * v * (1.0f + erff(v * 0.70710678118654752f));
}

// ---------------------------------------------------------------------------
// 1) f32 -> bf16 elementwise conversion
// ---------------------------------------------------------------------------
__global__ __launch_bounds__(256) void cvt_bf16_kernel(const float* __restrict__ in,
                                                       __bf16* __restrict__ out, int n) {
  int i = blockIdx.x * 256 + threadIdx.x;
  if (i < n) out[i] = (__bf16)in[i];
}

// ---------------------------------------------------------------------------
// 2) mod = silu(time_emb) @ Wmod + bmod     [4,1024] x [1024,6144] -> [4,6144]
// ---------------------------------------------------------------------------
__global__ __launch_bounds__(256) void mod_kernel(const float* __restrict__ temb,
                                                  const float* __restrict__ Wmod,
                                                  const float* __restrict__ bmod,
                                                  float* __restrict__ modb) {
  __shared__ float s[1024];
  int tid = threadIdx.x, b = blockIdx.y;
#pragma unroll
  for (int j = 0; j < 4; ++j) {
    float t = temb[b * 1024 + tid + 256 * j];
    s[tid + 256 * j] = t / (1.0f + __expf(-t));   // silu
  }
  __syncthreads();
  int n = blockIdx.x * 256 + tid;
  float acc = bmod[n];
  for (int k = 0; k < 1024; ++k) acc += s[k] * Wmod[(size_t)k * 6144 + n];
  modb[(size_t)b * 6144 + n] = acc;
}

// ---------------------------------------------------------------------------
// 3) h = LN(x)*w+b, then AdaLN modulate: h*(1+sc)+sh, store bf16.
//    One 256-thread workgroup per row (D = 1024).
// ---------------------------------------------------------------------------
__global__ __launch_bounds__(256) void ln_mod_kernel(const float* __restrict__ x,
                                                     const float* __restrict__ w,
                                                     const float* __restrict__ bb,
                                                     const float* __restrict__ modb,
                                                     int sh_off, int sc_off,
                                                     __bf16* __restrict__ out) {
  __shared__ float red[256];
  int row = blockIdx.x, b = row >> 10, tid = threadIdx.x;
  const float* xr = x + (size_t)row * 1024;
  float lx[4], s = 0.f, s2 = 0.f;
#pragma unroll
  for (int j = 0; j < 4; ++j) {
    lx[j] = xr[tid + 256 * j];
    s += lx[j]; s2 += lx[j] * lx[j];
  }
  red[tid] = s; __syncthreads();
  for (int st = 128; st > 0; st >>= 1) { if (tid < st) red[tid] += red[tid + st]; __syncthreads(); }
  float mean = red[0] * (1.0f / 1024.0f);
  __syncthreads();
  red[tid] = s2; __syncthreads();
  for (int st = 128; st > 0; st >>= 1) { if (tid < st) red[tid] += red[tid + st]; __syncthreads(); }
  float var = red[0] * (1.0f / 1024.0f) - mean * mean;
  float rs = rsqrtf(var + 1e-5f);
  const float* shp = modb + (size_t)b * 6144 + sh_off;
  const float* scp = modb + (size_t)b * 6144 + sc_off;
#pragma unroll
  for (int j = 0; j < 4; ++j) {
    int c = tid + 256 * j;
    float hn = (lx[j] - mean) * rs * w[c] + bb[c];
    out[(size_t)row * 1024 + c] = (__bf16)(hn * (1.0f + scp[c]) + shp[c]);
  }
}

// ---------------------------------------------------------------------------
// 4) Tiled WMMA GEMM: C[M,N] = A[M,K](bf16) @ B[K,N](bf16) + bias (+ epilogue)
//    Block tile 128x64, BK=64. 256 threads = 8 waves in a 4x2 grid; each wave
//    computes a 32x32 micro-tile = 8 WMMAs per K-stage (2 A-frags x 2 B-frags
//    x 2 k-steps). Register double-buffering hides global-load latency under
//    the WMMAs; global_prefetch_b8 pre-warms L2 two stages ahead.
// ---------------------------------------------------------------------------
enum { EPI_BF16 = 0, EPI_GELU = 1, EPI_RES = 2 };

template <int EPI>
__global__ __launch_bounds__(256) void gemm_bf16_kernel(
    const __bf16* __restrict__ A, int lda,
    const __bf16* __restrict__ B, int ldb,
    const float* __restrict__ bias,
    const float* __restrict__ res,
    void* __restrict__ outv, int ldo, int K) {
  constexpr int AS = 72;            // LDS k-stride (144B rows: 16B aligned, conflict-skewed)
  __shared__ __bf16 As[128 * AS];   // [m][k]
  __shared__ __bf16 Bs[64 * AS];    // transposed: [n][k]
  int tid = threadIdx.x;
  int w = tid >> 5, l = tid & 31, ln = l & 15, grp = l >> 4;
  int wm = w & 3, wn = w >> 2;                 // wave grid: 4 (M) x 2 (N)
  int mb = blockIdx.y * 128, nb = blockIdx.x * 64;
  int arow = tid >> 1, acol = (tid & 1) * 32;  // A tile 128x64: 32 elems/thread
  int brow = tid >> 2, bcol = (tid & 3) * 16;  // B tile 64x64:  16 elems/thread
  const __bf16* Ag = A + (size_t)(mb + arow) * lda + acol;
  const __bf16* Bg = B + (size_t)brow * ldb + nb + bcol;

  uint4 arg[4], brg[2];
#pragma unroll
  for (int j = 0; j < 4; ++j) arg[j] = *(const uint4*)(Ag + 8 * j);
#pragma unroll
  for (int j = 0; j < 2; ++j) brg[j] = *(const uint4*)(Bg + 8 * j);

  f32x8 acc[2][2] = {};
  for (int kb = 0; kb < K; kb += 64) {
    __syncthreads();                      // previous stage's LDS reads complete
#pragma unroll
    for (int j = 0; j < 4; ++j) *(uint4*)&As[arow * AS + acol + 8 * j] = arg[j];
#pragma unroll
    for (int part = 0; part < 2; ++part) {
      union { uint4 q; __bf16 h[8]; } u;
      u.q = brg[part];
#pragma unroll
      for (int j = 0; j < 8; ++j) Bs[(bcol + part * 8 + j) * AS + brow] = u.h[j];
    }
    __syncthreads();
    if (kb + 64 < K) {                    // fetch next stage under the WMMAs
#pragma unroll
      for (int j = 0; j < 4; ++j)
        arg[j] = *(const uint4*)(Ag + (kb + 64) + 8 * j);
#pragma unroll
      for (int j = 0; j < 2; ++j)
        brg[j] = *(const uint4*)(Bg + (size_t)(kb + 64) * ldb + 8 * j);
      if (kb + 128 < K) {                 // L2 prefetch two stages ahead
        __builtin_prefetch(Ag + kb + 128, 0, 0);
        __builtin_prefetch(Bg + (size_t)(kb + 128) * ldb, 0, 0);
      }
    }
#pragma unroll
    for (int kk = 0; kk < 64; kk += 32) {
      const __bf16* a0p = &As[(wm * 32 + ln) * AS + kk];
      const __bf16* a1p = &As[(wm * 32 + 16 + ln) * AS + kk];
      bf16x16 a0 = cat8(*(const bf16x8*)(a0p + grp * 16), *(const bf16x8*)(a0p + grp * 16 + 8));
      bf16x16 a1 = cat8(*(const bf16x8*)(a1p + grp * 16), *(const bf16x8*)(a1p + grp * 16 + 8));
      bf16x16 b0 = *(const bf16x16*)&Bs[(wn * 32 + ln) * AS + kk + grp * 16];
      bf16x16 b1 = *(const bf16x16*)&Bs[(wn * 32 + 16 + ln) * AS + kk + grp * 16];
      acc[0][0] = wmma_bf16(a0, b0, acc[0][0]);
      acc[0][1] = wmma_bf16(a0, b1, acc[0][1]);
      acc[1][0] = wmma_bf16(a1, b0, acc[1][0]);
      acc[1][1] = wmma_bf16(a1, b1, acc[1][1]);
    }
  }

  int c0 = nb + wn * 32 + ln, c1 = c0 + 16;
  float bias0 = bias[c0], bias1 = bias[c1];
#pragma unroll
  for (int sub = 0; sub < 2; ++sub) {
    int row0 = mb + wm * 32 + sub * 16 + grp * 8;
    if constexpr (EPI == EPI_RES) {
      float* out = (float*)outv;
#pragma unroll
      for (int i = 0; i < 8; ++i) {
        size_t r = (size_t)(row0 + i) * ldo;
        out[r + c0] = acc[sub][0][i] + bias0 + res[r + c0];
        out[r + c1] = acc[sub][1][i] + bias1 + res[r + c1];
      }
    } else {
      __bf16* out = (__bf16*)outv;
#pragma unroll
      for (int i = 0; i < 8; ++i) {
        float v0 = acc[sub][0][i] + bias0, v1 = acc[sub][1][i] + bias1;
        if constexpr (EPI == EPI_GELU) { v0 = gelu_exact(v0); v1 = gelu_exact(v1); }
        size_t r = (size_t)(row0 + i) * ldo;
        out[r + c0] = (__bf16)v0;
        out[r + c1] = (__bf16)v1;
      }
    }
  }
}

// ---------------------------------------------------------------------------
// 5) Vt[bh][f][nk] = V[b*Nk+nk][coff + h*64 + f]  (bf16 transpose for P@V)
// ---------------------------------------------------------------------------
__global__ __launch_bounds__(256) void transpose_v_kernel(const __bf16* __restrict__ v,
                                                          int ldv, int coff, int Nk,
                                                          __bf16* __restrict__ vt) {
  int bh = blockIdx.x, b = bh >> 4, h = bh & 15;
  int idx = blockIdx.y * 256 + threadIdx.x;    // over 64*Nk
  int f = idx / Nk, nk = idx - f * Nk;
  vt[(size_t)bh * 64 * Nk + (size_t)f * Nk + nk] =
      v[(size_t)(b * Nk + nk) * ldv + coff + h * 64 + f];
}

// ---------------------------------------------------------------------------
// 6) Flash attention, one wave32 per (batch*head, 16-query block).
//    S = Q K^T * scale (K features contiguous -> direct B-frag loads),
//    online softmax via 16x32 fp32 tile in LDS, P requantized to bf16,
//    O += P @ V using pre-transposed Vt (contiguous B-frag loads).
// ---------------------------------------------------------------------------
__global__ __launch_bounds__(32) void attn_kernel(
    const __bf16* __restrict__ Q, int ldq, int qoff,
    const __bf16* __restrict__ Kmat, int ldk, int koff,
    const __bf16* __restrict__ Vt, int Nk, float scale,
    __bf16* __restrict__ O, int ldo) {
  __shared__ float  Sf[16][32];
  __shared__ __bf16 Pl[16][32];
  __shared__ float  corrs[16];
  __shared__ float  invl[16];

  int bh = blockIdx.x, b = bh >> 4, h = bh & 15;
  int qb = blockIdx.y;
  int l = threadIdx.x, ln = l & 15, grp = l >> 4;
  int qrow0 = b * 1024 + qb * 16;

  const __bf16* qp = Q + (size_t)(qrow0 + ln) * ldq + qoff + h * 64;
  bf16x16 qa0 = cat8(*(const bf16x8*)(qp + grp * 16), *(const bf16x8*)(qp + grp * 16 + 8));
  bf16x16 qa1 = cat8(*(const bf16x8*)(qp + 32 + grp * 16), *(const bf16x8*)(qp + 32 + grp * 16 + 8));

  const __bf16* kp  = Kmat + koff + h * 64;
  const __bf16* vtp = Vt + (size_t)bh * 64 * Nk;

  f32x8 o[4] = {};
  float m_r = -3.0e38f, l_r = 0.0f;

  for (int kc = 0; kc < Nk; kc += 32) {
    const __bf16* kr0 = kp + (size_t)(b * Nk + kc + ln) * ldk;
    const __bf16* kr1 = kp + (size_t)(b * Nk + kc + 16 + ln) * ldk;
    f32x8 s0 = {}, s1 = {};
    s0 = wmma_bf16(qa0, *(const bf16x16*)(kr0 + grp * 16), s0);
    s0 = wmma_bf16(qa1, *(const bf16x16*)(kr0 + 32 + grp * 16), s0);
    s1 = wmma_bf16(qa0, *(const bf16x16*)(kr1 + grp * 16), s1);
    s1 = wmma_bf16(qa1, *(const bf16x16*)(kr1 + 32 + grp * 16), s1);
#pragma unroll
    for (int i = 0; i < 8; ++i) {
      Sf[grp * 8 + i][ln]      = s0[i] * scale;
      Sf[grp * 8 + i][16 + ln] = s1[i] * scale;
    }
    __syncthreads();
    if (l < 16) {
      float tm = -3.0e38f;
#pragma unroll
      for (int j = 0; j < 32; ++j) tm = fmaxf(tm, Sf[l][j]);
      float nm = fmaxf(m_r, tm);
      float corr = __expf(m_r - nm);
      float sum = 0.0f;
#pragma unroll
      for (int j = 0; j < 32; ++j) {
        float p = __expf(Sf[l][j] - nm);
        sum += p;
        Pl[l][j] = (__bf16)p;
      }
      l_r = l_r * corr + sum;
      m_r = nm;
      corrs[l] = corr;
    }
    __syncthreads();
    float cf[8];
#pragma unroll
    for (int i = 0; i < 8; ++i) cf[i] = corrs[grp * 8 + i];
#pragma unroll
    for (int t = 0; t < 4; ++t)
#pragma unroll
      for (int i = 0; i < 8; ++i) o[t][i] *= cf[i];

    bf16x16 pa = cat8(*(const bf16x8*)&Pl[ln][grp * 16], *(const bf16x8*)&Pl[ln][grp * 16 + 8]);
    const __bf16* vb = vtp + kc + grp * 16;
#pragma unroll
    for (int t = 0; t < 4; ++t)
      o[t] = wmma_bf16(pa, *(const bf16x16*)(vb + (size_t)(t * 16 + ln) * Nk), o[t]);
    __syncthreads();
  }

  if (l < 16) invl[l] = 1.0f / l_r;
  __syncthreads();
  float il[8];
#pragma unroll
  for (int i = 0; i < 8; ++i) il[i] = invl[grp * 8 + i];
  __bf16* op = O + (size_t)qrow0 * ldo + h * 64;
#pragma unroll
  for (int t = 0; t < 4; ++t)
#pragma unroll
    for (int i = 0; i < 8; ++i)
      op[(size_t)(grp * 8 + i) * ldo + t * 16 + ln] = (__bf16)(o[t][i] * il[i]);
}

// ---------------------------------------------------------------------------
// Host orchestration
// ---------------------------------------------------------------------------
extern "C" void kernel_launch(void* const* d_in, const int* in_sizes, int n_in,
                              void* d_out, int out_size, void* d_ws, size_t ws_size,
                              hipStream_t stream) {
  (void)in_sizes; (void)n_in; (void)out_size; (void)ws_size;
  const float* x    = (const float*)d_in[0];
  const float* ctx  = (const float*)d_in[1];
  const float* temb = (const float*)d_in[2];
  const float* ln1w = (const float*)d_in[3];  const float* ln1b = (const float*)d_in[4];
  const float* ln2w = (const float*)d_in[5];  const float* ln2b = (const float*)d_in[6];
  const float* ln3w = (const float*)d_in[7];  const float* ln3b = (const float*)d_in[8];
  const float* Wqkv = (const float*)d_in[9];  const float* bqkv = (const float*)d_in[10];
  const float* Wos  = (const float*)d_in[11]; const float* bos  = (const float*)d_in[12];
  const float* Wq   = (const float*)d_in[13]; const float* bq   = (const float*)d_in[14];
  const float* Wk   = (const float*)d_in[15]; const float* bk   = (const float*)d_in[16];
  const float* Wv   = (const float*)d_in[17]; const float* bv   = (const float*)d_in[18];
  const float* Woc  = (const float*)d_in[19]; const float* boc  = (const float*)d_in[20];
  const float* W1   = (const float*)d_in[21]; const float* b1   = (const float*)d_in[22];
  const float* W2   = (const float*)d_in[23]; const float* b2   = (const float*)d_in[24];
  const float* Wmod = (const float*)d_in[25]; const float* bmod = (const float*)d_in[26];
  float* out = (float*)d_out;

  // ---- workspace carve (256B aligned) ----
  char* p = (char*)d_ws;
  auto carve = [&](size_t bytes) -> void* {
    void* r = (void*)p; p += (bytes + 255) & ~(size_t)255; return r;
  };
  __bf16* wqkv_b = (__bf16*)carve((size_t)1024 * 3072 * 2);
  __bf16* wos_b  = (__bf16*)carve((size_t)1024 * 1024 * 2);
  __bf16* wq_b   = (__bf16*)carve((size_t)1024 * 1024 * 2);
  __bf16* wk_b   = (__bf16*)carve((size_t)768  * 1024 * 2);
  __bf16* wv_b   = (__bf16*)carve((size_t)768  * 1024 * 2);
  __bf16* woc_b  = (__bf16*)carve((size_t)1024 * 1024 * 2);
  __bf16* w1_b   = (__bf16*)carve((size_t)1024 * 4096 * 2);
  __bf16* w2_b   = (__bf16*)carve((size_t)4096 * 1024 * 2);
  __bf16* ctx_b  = (__bf16*)carve((size_t)4 * 256 * 768 * 2);
  float*  modb   = (float*) carve((size_t)4 * 6144 * 4);
  __bf16* hbuf   = (__bf16*)carve((size_t)4096 * 1024 * 2);
  __bf16* qkv    = (__bf16*)carve((size_t)4096 * 3072 * 2);
  __bf16* vt     = (__bf16*)carve((size_t)64 * 64 * 1024 * 2);
  __bf16* attn_o = (__bf16*)carve((size_t)4096 * 1024 * 2);
  __bf16* q2     = (__bf16*)carve((size_t)4096 * 1024 * 2);
  __bf16* k2     = (__bf16*)carve((size_t)1024 * 1024 * 2);
  __bf16* v2     = (__bf16*)carve((size_t)1024 * 1024 * 2);
  __bf16* ff     = (__bf16*)carve((size_t)4096 * 4096 * 2);

  auto cvt = [&](const float* src, __bf16* dst, int n) {
    cvt_bf16_kernel<<<(n + 255) / 256, 256, 0, stream>>>(src, dst, n);
  };
  cvt(Wqkv, wqkv_b, 1024 * 3072);
  cvt(Wos,  wos_b,  1024 * 1024);
  cvt(Wq,   wq_b,   1024 * 1024);
  cvt(Wk,   wk_b,   768  * 1024);
  cvt(Wv,   wv_b,   768  * 1024);
  cvt(Woc,  woc_b,  1024 * 1024);
  cvt(W1,   w1_b,   1024 * 4096);
  cvt(W2,   w2_b,   4096 * 1024);
  cvt(ctx,  ctx_b,  4 * 256 * 768);

  // modulation
  mod_kernel<<<dim3(24, 4), 256, 0, stream>>>(temb, Wmod, bmod, modb);

  // ===== self-attention branch =====
  ln_mod_kernel<<<4096, 256, 0, stream>>>(x, ln1w, ln1b, modb, 0, 1024, hbuf);
  gemm_bf16_kernel<EPI_BF16><<<dim3(48, 32), 256, 0, stream>>>(
      hbuf, 1024, wqkv_b, 3072, bqkv, nullptr, qkv, 3072, 1024);
  transpose_v_kernel<<<dim3(64, 256), 256, 0, stream>>>(qkv, 3072, 2048, 1024, vt);
  attn_kernel<<<dim3(64, 64), 32, 0, stream>>>(
      qkv, 3072, 0, qkv, 3072, 1024, vt, 1024, 0.125f, attn_o, 1024);
  gemm_bf16_kernel<EPI_RES><<<dim3(16, 32), 256, 0, stream>>>(
      attn_o, 1024, wos_b, 1024, bos, x, out, 1024, 1024);

  // ===== cross-attention branch =====
  ln_mod_kernel<<<4096, 256, 0, stream>>>(out, ln2w, ln2b, modb, 2048, 3072, hbuf);
  gemm_bf16_kernel<EPI_BF16><<<dim3(16, 32), 256, 0, stream>>>(
      hbuf, 1024, wq_b, 1024, bq, nullptr, q2, 1024, 1024);
  gemm_bf16_kernel<EPI_BF16><<<dim3(16, 8), 256, 0, stream>>>(
      ctx_b, 768, wk_b, 1024, bk, nullptr, k2, 1024, 768);
  gemm_bf16_kernel<EPI_BF16><<<dim3(16, 8), 256, 0, stream>>>(
      ctx_b, 768, wv_b, 1024, bv, nullptr, v2, 1024, 768);
  transpose_v_kernel<<<dim3(64, 64), 256, 0, stream>>>(v2, 1024, 0, 256, vt);
  attn_kernel<<<dim3(64, 64), 32, 0, stream>>>(
      q2, 1024, 0, k2, 1024, 0, vt, 256, 0.125f, attn_o, 1024);
  gemm_bf16_kernel<EPI_RES><<<dim3(16, 32), 256, 0, stream>>>(
      attn_o, 1024, woc_b, 1024, boc, out, out, 1024, 1024);

  // ===== feed-forward branch =====
  ln_mod_kernel<<<4096, 256, 0, stream>>>(out, ln3w, ln3b, modb, 4096, 5120, hbuf);
  gemm_bf16_kernel<EPI_GELU><<<dim3(64, 32), 256, 0, stream>>>(
      hbuf, 1024, w1_b, 4096, b1, nullptr, ff, 4096, 1024);
  gemm_bf16_kernel<EPI_RES><<<dim3(16, 32), 256, 0, stream>>>(
      ff, 4096, w2_b, 1024, b2, out, out, 1024, 4096);
}
